// AverageAttention_9191230013963
// MI455X (gfx1250) — matrix operand order repaired and checked by reading, not compiled, not run on hardware
//
#include <hip/hip_runtime.h>

// ---------------------------------------------------------------------------
// AverageAttention block: cumulative-average -> LayerNorm -> FFN(+res) -> gate
// B=8, L=2048, D=1024.  GEMMs run on v_wmma_f32_16x16x32_f16 (f32 accum),
// double-buffered LDS pipeline, async global->LDS DMA (ASYNCcnt-tracked).
// ---------------------------------------------------------------------------

typedef _Float16 half_t;
typedef __attribute__((ext_vector_type(16))) _Float16 v16h;
typedef __attribute__((ext_vector_type(8)))  float    v8f;
typedef __attribute__((ext_vector_type(4)))  float    v4f;

#define DIM     1024
#define SEQ_B   8
#define SEQ_L   2048
#define M_TOT   (SEQ_B * SEQ_L)   // 16384 rows

// GEMM tiling: 128x64 block tile, K-chunks of 32, 8 wave32s per block.
// Waves tile the block 4x2; each wave owns a 32x32 patch (2x2 WMMA tiles).
#define BM      128
#define BN      64
#define BK      32
#define NTHREADS 256

#if __has_builtin(__builtin_amdgcn_global_load_async_to_lds_b128) && \
    __has_builtin(__builtin_amdgcn_s_wait_asynccnt)
#define HAS_ASYNC_LDS 1
#else
#define HAS_ASYNC_LDS 0
#endif

// ------------------------------ fragments ----------------------------------

union FragH { v16h v; v4f q[2]; };

// A fragment: 16x32 f16 tile, sA layout [BM][BK] row-major.
// lane (l&15) = row M; half-wave c selects K runs [8c..8c+7], [8c+16..8c+23].
__device__ inline v16h load_frag_a(const half_t* sA, int lane, int mBase) {
    const int row = mBase + (lane & 15);
    const int kb  = (lane >> 4) * 8;
    FragH f;
    f.q[0] = *reinterpret_cast<const v4f*>(sA + row * BK + kb);
    f.q[1] = *reinterpret_cast<const v4f*>(sA + row * BK + kb + 16);
    return f.v;
}

// B fragment: 32x16 f16 tile, sB layout [BN][BK] (per-column K contiguous).
// lane (l&15) = column N; half-wave c selects K runs [8c..8c+7], [8c+16..8c+23].
__device__ inline v16h load_frag_b(const half_t* sB, int lane, int nBase) {
    const int col = nBase + (lane & 15);
    const int kb  = (lane >> 4) * 8;
    FragH f;
    f.q[0] = *reinterpret_cast<const v4f*>(sB + col * BK + kb);
    f.q[1] = *reinterpret_cast<const v4f*>(sB + col * BK + kb + 16);
    return f.v;
}

__device__ inline v8f wmma_f16(v16h a, v16h b, v8f c) {
    return __builtin_amdgcn_wmma_f32_16x16x32_f16(
        false, a, false, b, (short)0, c, false, false);
}

#if HAS_ASYNC_LDS
// Builtin signature (from hipcc diagnostic):
//   (int __vector(4) __device__ *src, int __vector(4) __shared__ *dst,
//    imm offset, imm cpol)
typedef int v4i_t __attribute__((__vector_size__(16)));
typedef __attribute__((address_space(1))) v4i_t* gvec_p;
typedef __attribute__((address_space(3))) v4i_t* lvec_p;

__device__ inline void async_b128(const half_t* g, half_t* l) {
    __builtin_amdgcn_global_load_async_to_lds_b128(
        (gvec_p)(half_t*)g, (lvec_p)l, 0, 0);
}
#endif

// --------------------------- stage 1: cumulative mean -----------------------

__global__ void __launch_bounds__(256)
k_cumavg(const float* __restrict__ in, float* __restrict__ avg) {
    const int idx = blockIdx.x * 256 + threadIdx.x;   // 0 .. B*D-1
    const int b = idx / DIM;
    const int d = idx % DIM;
    const float* src = in  + (size_t)b * SEQ_L * DIM + d;
    float*       dst = avg + (size_t)b * SEQ_L * DIM + d;
    float s = 0.0f;
    for (int t = 0; t < SEQ_L; ++t) {
        s += src[(size_t)t * DIM];
        dst[(size_t)t * DIM] = s / (float)(t + 1);
    }
}

// --------------------------- stage 2: LayerNorm -> f16 ----------------------

__global__ void __launch_bounds__(256)
k_layernorm(const float* __restrict__ avg, const float* __restrict__ g,
            const float* __restrict__ bta, half_t* __restrict__ xh) {
    __shared__ float red[256];
    const int row = blockIdx.x;
    const int tid = threadIdx.x;
    const float* src = avg + (size_t)row * DIM;

    float s = 0.0f, ss = 0.0f;
    for (int d = tid; d < DIM; d += 256) {
        float v = src[d];
        s  += v;
        ss += v * v;
    }
    red[tid] = s;  __syncthreads();
    for (int off = 128; off > 0; off >>= 1) {
        if (tid < off) red[tid] += red[tid + off];
        __syncthreads();
    }
    const float mean = red[0] * (1.0f / DIM);
    __syncthreads();
    red[tid] = ss; __syncthreads();
    for (int off = 128; off > 0; off >>= 1) {
        if (tid < off) red[tid] += red[tid + off];
        __syncthreads();
    }
    const float var  = red[0] * (1.0f / DIM) - mean * mean;
    const float rstd = rsqrtf(var + 1e-6f);

    half_t* dst = xh + (size_t)row * DIM;
    for (int d = tid; d < DIM; d += 256) {
        float v = (src[d] - mean) * rstd * g[d] + bta[d];
        dst[d] = (half_t)v;
    }
}

// --------------------------- conversion helpers -----------------------------

__global__ void __launch_bounds__(256)
k_cvt_f16(const float* __restrict__ src, half_t* __restrict__ dst, int n) {
    const int i = blockIdx.x * 256 + threadIdx.x;
    if (i < n) dst[i] = (half_t)src[i];
}

// dst[n*K + k] = (f16) src[k*N + n]   (weight transpose: B columns contiguous)
__global__ void __launch_bounds__(256)
k_cvt_tr(const float* __restrict__ src, half_t* __restrict__ dst, int K, int N) {
    const long i = (long)blockIdx.x * 256 + threadIdx.x;
    if (i < (long)K * N) {
        const int n = (int)(i / K);
        const int k = (int)(i % K);
        dst[i] = (half_t)src[(size_t)k * N + n];
    }
}

// --------------------------- tiled WMMA GEMM --------------------------------
// MODE 0: outH = f16(relu(A @ Wt^T + bias))
// MODE 1: v = A @ Wt^T + bias + resid ; outF = v ; outH = f16(v)
template <int MODE>
__global__ void __launch_bounds__(NTHREADS)
k_gemm(const half_t* __restrict__ A, const half_t* __restrict__ Wt,
       const float* __restrict__ bias, const float* __restrict__ resid,
       float* __restrict__ outF, half_t* __restrict__ outH,
       int K, int N) {
    __shared__ half_t sA[2][BM * BK];
    __shared__ half_t sB[2][BN * BK];

    const int tid  = threadIdx.x;
    const int lane = tid & 31;
    const int wave = tid >> 5;
    const int wm   = (wave & 3) * 32;         // wave row base within block
    const int wn   = (wave >> 2) * 32;        // wave col base within block
    const int m0   = blockIdx.y * BM;
    const int n0   = blockIdx.x * BN;

    v8f acc[2][2];
    for (int i = 0; i < 2; ++i)
        for (int j = 0; j < 2; ++j)
            for (int e = 0; e < 8; ++e) acc[i][j][e] = 0.0f;

    // global->LDS copy assignments (16B vector pieces)
    const int ar = tid >> 1;                  // 0..127
    const int ak = (tid & 1) * 16;            // 0 / 16
    const int bn = tid >> 2;                  // 0..63
    const int bk = (tid & 3) * 8;             // 0/8/16/24

    const half_t* gA = A  + (size_t)(m0 + ar) * K + ak;
    const half_t* gB = Wt + (size_t)(n0 + bn) * K + bk;
    const int T = K / BK;
    int buf = 0;

#if HAS_ASYNC_LDS
    async_b128(gA,     &sA[0][ar * BK + ak]);
    async_b128(gA + 8, &sA[0][ar * BK + ak + 8]);
    async_b128(gB,     &sB[0][bn * BK + bk]);
    __builtin_amdgcn_s_wait_asynccnt(0);
    __syncthreads();
#else
    v4f rA0 = *reinterpret_cast<const v4f*>(gA);
    v4f rA1 = *reinterpret_cast<const v4f*>(gA + 8);
    v4f rB0 = *reinterpret_cast<const v4f*>(gB);
    *reinterpret_cast<v4f*>(&sA[0][ar * BK + ak])     = rA0;
    *reinterpret_cast<v4f*>(&sA[0][ar * BK + ak + 8]) = rA1;
    *reinterpret_cast<v4f*>(&sB[0][bn * BK + bk])     = rB0;
    __syncthreads();
#endif

    for (int t = 0; t < T; ++t) {
        const bool more = (t + 1 < T);
#if HAS_ASYNC_LDS
        if (more) {                           // DMA next tile into other buffer
            const half_t* nA = gA + (size_t)(t + 1) * BK;
            const half_t* nB = gB + (size_t)(t + 1) * BK;
            async_b128(nA,     &sA[buf ^ 1][ar * BK + ak]);
            async_b128(nA + 8, &sA[buf ^ 1][ar * BK + ak + 8]);
            async_b128(nB,     &sB[buf ^ 1][bn * BK + bk]);
        }
#else
        if (more) {                           // register-staged prefetch
            const half_t* nA = gA + (size_t)(t + 1) * BK;
            const half_t* nB = gB + (size_t)(t + 1) * BK;
            rA0 = *reinterpret_cast<const v4f*>(nA);
            rA1 = *reinterpret_cast<const v4f*>(nA + 8);
            rB0 = *reinterpret_cast<const v4f*>(nB);
        }
#endif
        // Load every fragment into its own registers, then issue the WMMAs
        // back-to-back so LDS latency is paid once per K-step.
        const half_t* pA = sA[buf];
        const half_t* pB = sB[buf];
        const v16h a0 = load_frag_a(pA, lane, wm);
        const v16h a1 = load_frag_a(pA, lane, wm + 16);
        const v16h b0 = load_frag_b(pB, lane, wn);
        const v16h b1 = load_frag_b(pB, lane, wn + 16);
        acc[0][0] = wmma_f16(a0, b0, acc[0][0]);
        acc[0][1] = wmma_f16(a0, b1, acc[0][1]);
        acc[1][0] = wmma_f16(a1, b0, acc[1][0]);
        acc[1][1] = wmma_f16(a1, b1, acc[1][1]);

        if (more) {
#if HAS_ASYNC_LDS
            __builtin_amdgcn_s_wait_asynccnt(0);
            __syncthreads();                  // publish prefetched tile
#else
            *reinterpret_cast<v4f*>(&sA[buf ^ 1][ar * BK + ak])     = rA0;
            *reinterpret_cast<v4f*>(&sA[buf ^ 1][ar * BK + ak + 8]) = rA1;
            *reinterpret_cast<v4f*>(&sB[buf ^ 1][bn * BK + bk])     = rB0;
            __syncthreads();
#endif
        }
        buf ^= 1;
    }

    // Epilogue.  C/D layout: VGPR r holds M = 8*(lane>=16)+r, N = lane&15.
    const int mo   = (lane >> 4) * 8;
    const int col0 = lane & 15;
    #pragma unroll
    for (int j = 0; j < 2; ++j) {
        const int col = n0 + wn + j * 16 + col0;
        const float bv = bias[col];
        #pragma unroll
        for (int i = 0; i < 2; ++i) {
            #pragma unroll
            for (int r = 0; r < 8; ++r) {
                const int row = m0 + wm + i * 16 + mo + r;
                float v = acc[i][j][r] + bv;
                if constexpr (MODE == 0) {
                    v = fmaxf(v, 0.0f);
                    outH[(size_t)row * N + col] = (half_t)v;
                } else {
                    v += resid[(size_t)row * N + col];
                    outF[(size_t)row * N + col] = v;
                    outH[(size_t)row * N + col] = (half_t)v;
                }
            }
        }
    }
}

// --------------------------- fused gating GEMM ------------------------------
// Block computes gating columns [n0,n0+64) (input gate) and [n0+D,n0+D+64)
// (forget gate) with one shared A tile, then writes
// out = sigmoid(ig)*inputs + sigmoid(fg)*ffn directly.  A = concat(inputs_f16,
// ffn_f16) handled by switching the A pointer at k = D (BK divides D).
__global__ void __launch_bounds__(NTHREADS)
k_gate(const half_t* __restrict__ Ain, const half_t* __restrict__ Affn,
       const half_t* __restrict__ Gwt, const float* __restrict__ gb,
       const float* __restrict__ inputs, const float* __restrict__ ffn,
       float* __restrict__ out) {
    __shared__ half_t sA [2][BM * BK];
    __shared__ half_t sBi[2][BN * BK];
    __shared__ half_t sBf[2][BN * BK];

    const int K2   = 2 * DIM;                 // 2048
    const int tid  = threadIdx.x;
    const int lane = tid & 31;
    const int wave = tid >> 5;
    const int wm   = (wave & 3) * 32;
    const int wn   = (wave >> 2) * 32;
    const int m0   = blockIdx.y * BM;
    const int n0   = blockIdx.x * BN;         // n0 in [0, DIM)

    v8f accI[2][2], accF[2][2];
    for (int i = 0; i < 2; ++i)
        for (int j = 0; j < 2; ++j)
            for (int e = 0; e < 8; ++e) { accI[i][j][e] = 0.0f; accF[i][j][e] = 0.0f; }

    const int ar = tid >> 1;
    const int ak = (tid & 1) * 16;
    const int bn = tid >> 2;
    const int bk = (tid & 3) * 8;

    auto aPtr = [&](int t) -> const half_t* {
        const int k0 = t * BK;
        return (k0 < DIM) ? Ain  + (size_t)(m0 + ar) * DIM + k0 + ak
                          : Affn + (size_t)(m0 + ar) * DIM + (k0 - DIM) + ak;
    };
    const half_t* gBi = Gwt + (size_t)(n0 + bn) * K2 + bk;
    const half_t* gBf = Gwt + (size_t)(n0 + DIM + bn) * K2 + bk;
    const int T = K2 / BK;
    int buf = 0;

#if HAS_ASYNC_LDS
    async_b128(aPtr(0),     &sA[0][ar * BK + ak]);
    async_b128(aPtr(0) + 8, &sA[0][ar * BK + ak + 8]);
    async_b128(gBi,         &sBi[0][bn * BK + bk]);
    async_b128(gBf,         &sBf[0][bn * BK + bk]);
    __builtin_amdgcn_s_wait_asynccnt(0);
    __syncthreads();
#else
    v4f rA0 = *reinterpret_cast<const v4f*>(aPtr(0));
    v4f rA1 = *reinterpret_cast<const v4f*>(aPtr(0) + 8);
    v4f rBi = *reinterpret_cast<const v4f*>(gBi);
    v4f rBf = *reinterpret_cast<const v4f*>(gBf);
    *reinterpret_cast<v4f*>(&sA[0][ar * BK + ak])     = rA0;
    *reinterpret_cast<v4f*>(&sA[0][ar * BK + ak + 8]) = rA1;
    *reinterpret_cast<v4f*>(&sBi[0][bn * BK + bk])    = rBi;
    *reinterpret_cast<v4f*>(&sBf[0][bn * BK + bk])    = rBf;
    __syncthreads();
#endif

    for (int t = 0; t < T; ++t) {
        const bool more = (t + 1 < T);
#if HAS_ASYNC_LDS
        if (more) {
            const half_t* nA = aPtr(t + 1);
            async_b128(nA,                           &sA [buf ^ 1][ar * BK + ak]);
            async_b128(nA + 8,                       &sA [buf ^ 1][ar * BK + ak + 8]);
            async_b128(gBi + (size_t)(t + 1) * BK,   &sBi[buf ^ 1][bn * BK + bk]);
            async_b128(gBf + (size_t)(t + 1) * BK,   &sBf[buf ^ 1][bn * BK + bk]);
        }
#else
        if (more) {
            const half_t* nA = aPtr(t + 1);
            rA0 = *reinterpret_cast<const v4f*>(nA);
            rA1 = *reinterpret_cast<const v4f*>(nA + 8);
            rBi = *reinterpret_cast<const v4f*>(gBi + (size_t)(t + 1) * BK);
            rBf = *reinterpret_cast<const v4f*>(gBf + (size_t)(t + 1) * BK);
        }
#endif
        const half_t* pA = sA[buf];
        const v16h a0  = load_frag_a(pA, lane, wm);
        const v16h a1  = load_frag_a(pA, lane, wm + 16);
        const v16h bi0 = load_frag_b(sBi[buf], lane, wn);
        const v16h bi1 = load_frag_b(sBi[buf], lane, wn + 16);
        const v16h bf0 = load_frag_b(sBf[buf], lane, wn);
        const v16h bf1 = load_frag_b(sBf[buf], lane, wn + 16);
        accI[0][0] = wmma_f16(a0, bi0, accI[0][0]);
        accI[0][1] = wmma_f16(a0, bi1, accI[0][1]);
        accI[1][0] = wmma_f16(a1, bi0, accI[1][0]);
        accI[1][1] = wmma_f16(a1, bi1, accI[1][1]);
        accF[0][0] = wmma_f16(a0, bf0, accF[0][0]);
        accF[0][1] = wmma_f16(a0, bf1, accF[0][1]);
        accF[1][0] = wmma_f16(a1, bf0, accF[1][0]);
        accF[1][1] = wmma_f16(a1, bf1, accF[1][1]);

        if (more) {
#if HAS_ASYNC_LDS
            __builtin_amdgcn_s_wait_asynccnt(0);
            __syncthreads();
#else
            *reinterpret_cast<v4f*>(&sA [buf ^ 1][ar * BK + ak])     = rA0;
            *reinterpret_cast<v4f*>(&sA [buf ^ 1][ar * BK + ak + 8]) = rA1;
            *reinterpret_cast<v4f*>(&sBi[buf ^ 1][bn * BK + bk])     = rBi;
            *reinterpret_cast<v4f*>(&sBf[buf ^ 1][bn * BK + bk])     = rBf;
            __syncthreads();
#endif
        }
        buf ^= 1;
    }

    const int mo   = (lane >> 4) * 8;
    const int col0 = lane & 15;
    #pragma unroll
    for (int j = 0; j < 2; ++j) {
        const int col = n0 + wn + j * 16 + col0;
        const float bI = gb[col];
        const float bF = gb[col + DIM];
        #pragma unroll
        for (int i = 0; i < 2; ++i) {
            #pragma unroll
            for (int r = 0; r < 8; ++r) {
                const int row = m0 + wm + i * 16 + mo + r;
                const size_t o = (size_t)row * DIM + col;
                const float gi = accI[i][j][r] + bI;
                const float gf = accF[i][j][r] + bF;
                const float si = 1.0f / (1.0f + expf(-gi));
                const float sf = 1.0f / (1.0f + expf(-gf));
                out[o] = si * inputs[o] + sf * ffn[o];
            }
        }
    }
}

// --------------------------------- launch -----------------------------------

extern "C" void kernel_launch(void* const* d_in, const int* in_sizes, int n_in,
                              void* d_out, int out_size, void* d_ws, size_t ws_size,
                              hipStream_t stream) {
    const float* inputs = (const float*)d_in[0];
    const float* ln_g   = (const float*)d_in[1];
    const float* ln_b   = (const float*)d_in[2];
    const float* w1     = (const float*)d_in[3];
    const float* b1     = (const float*)d_in[4];
    const float* w2     = (const float*)d_in[5];
    const float* b2     = (const float*)d_in[6];
    const float* gw     = (const float*)d_in[7];
    const float* gb     = (const float*)d_in[8];

    float* out = (float*)d_out;                          // [M, D]
    float* ffn = out + (size_t)M_TOT * DIM;              // [M, D] (2nd output)

    // workspace carve-up
    char* ws = (char*)d_ws;
    size_t off = 0;
    float*  avg = (float*) (ws + off); off += (size_t)M_TOT * DIM * sizeof(float);
    half_t* xh  = (half_t*)(ws + off); off += (size_t)M_TOT * DIM * sizeof(half_t);
    half_t* hh  = (half_t*)(ws + off); off += (size_t)M_TOT * DIM * sizeof(half_t);
    half_t* inh = (half_t*)(ws + off); off += (size_t)M_TOT * DIM * sizeof(half_t);
    half_t* fh  = (half_t*)(ws + off); off += (size_t)M_TOT * DIM * sizeof(half_t);
    half_t* w1t = (half_t*)(ws + off); off += (size_t)DIM * DIM * sizeof(half_t);
    half_t* w2t = (half_t*)(ws + off); off += (size_t)DIM * DIM * sizeof(half_t);
    half_t* gwt = (half_t*)(ws + off); off += (size_t)4 * DIM * DIM * sizeof(half_t);
    (void)ws_size; (void)n_in; (void)in_sizes; (void)out_size;

    // weight convert+transpose (f32 -> f16, column-contiguous for WMMA B)
    k_cvt_tr<<<(DIM * DIM) / 256, 256, 0, stream>>>(w1, w1t, DIM, DIM);
    k_cvt_tr<<<(DIM * DIM) / 256, 256, 0, stream>>>(w2, w2t, DIM, DIM);
    k_cvt_tr<<<(4 * DIM * DIM) / 256, 256, 0, stream>>>(gw, gwt, 2 * DIM, 2 * DIM);
    k_cvt_f16<<<(M_TOT * DIM) / 256, 256, 0, stream>>>(inputs, inh, M_TOT * DIM);

    // cumulative average (prefix scan, coalesced across d)
    k_cumavg<<<(SEQ_B * DIM) / 256, 256, 0, stream>>>(inputs, avg);

    // LayerNorm -> f16 activations
    k_layernorm<<<M_TOT, 256, 0, stream>>>(avg, ln_g, ln_b, xh);

    // FFN: h = relu(x@w1 + b1); ffn = h@w2 + b2 + avg
    dim3 g1(DIM / BN, M_TOT / BM);
    k_gemm<0><<<g1, NTHREADS, 0, stream>>>(xh, w1t, b1, nullptr, nullptr, hh, DIM, DIM);
    k_gemm<1><<<g1, NTHREADS, 0, stream>>>(hh, w2t, b2, avg, ffn, fh, DIM, DIM);

    // fused gating: out = sig(concat@gw[:, :D])*inputs + sig(concat@gw[:, D:])*ffn
    dim3 g2(DIM / BN, M_TOT / BM);
    k_gate<<<g2, NTHREADS, 0, stream>>>(inh, fh, gwt, gb, inputs, ffn, out);
}
